// CA_model_58171037057363
// MI455X (gfx1250) — matrix operand order repaired and checked by reading, not compiled, not run on hardware
//
#include <hip/hip_runtime.h>

// ---------------------------------------------------------------------------
// CDNA5 (gfx1250) cross-attention model.
// f16 WMMA (f32 accumulate), f16 intermediates, double-buffered async
// global->LDS staging (ASYNCcnt partial waits), permlane16 softmax reductions,
// exp2-domain online softmax. wave32, 256-thread (8-wave) blocks.
// ---------------------------------------------------------------------------

typedef __attribute__((ext_vector_type(16))) _Float16 v16h;
typedef __attribute__((ext_vector_type(8)))  float    v8f;

#define D_MODEL 256
#define SEQ     512
#define NHEAD   4
#define HDIM    64
#define BATCH   64
#define MTOT    (BATCH * SEQ)   // 32768 rows

__device__ __forceinline__ v8f wmma_f16(v16h a, v16h b, v8f c) {
  return __builtin_amdgcn_wmma_f32_16x16x32_f16(
      false, a, false, b, (short)0, c, false, false);
}

// ---- fast exp2 (v_exp_f32 is natively exp2) -------------------------------
#if defined(__has_builtin)
#if __has_builtin(__builtin_amdgcn_exp2f)
#define HAS_EXP2 1
#endif
#endif
__device__ __forceinline__ float fast_exp2(float x) {
#ifdef HAS_EXP2
  return __builtin_amdgcn_exp2f(x);
#else
  return exp2f(x);
#endif
}

// ---- CDNA5 async global->LDS copies (ASYNCcnt) ----------------------------
__device__ __forceinline__ unsigned lds_off_u32(const void* p) {
  return (unsigned)(size_t)p;   // generic LDS ptr: low 32 bits = LDS offset
}
__device__ __forceinline__ void async_ld_b128(unsigned lds, unsigned goff,
                                              const void* base) {
  asm volatile("global_load_async_to_lds_b128 %0, %1, %2"
               :: "v"(lds), "v"(goff), "s"(base) : "memory");
}
__device__ __forceinline__ void async_ld_b64(unsigned lds, unsigned goff,
                                             const void* base) {
  asm volatile("global_load_async_to_lds_b64 %0, %1, %2"
               :: "v"(lds), "v"(goff), "s"(base) : "memory");
}
template <int N>
__device__ __forceinline__ void wait_async_le() {
  asm volatile("s_wait_asynccnt %0" :: "i"(N) : "memory");
}

// ---- wave32 16-lane-half reductions: v_permlane16_b32 ---------------------
#if defined(__has_builtin)
#if __has_builtin(__builtin_amdgcn_permlane16)
#define HAS_PL16 1
#endif
#endif

__device__ __forceinline__ float pl16(float v, unsigned s0, unsigned s1) {
#ifdef HAS_PL16
  union { float f; unsigned u; } a, r;
  a.f = v;
  r.u = __builtin_amdgcn_permlane16(a.u, a.u, s0, s1, false, false);
  return r.f;
#else
  (void)s0; (void)s1;
  return v;
#endif
}
__device__ __forceinline__ float red16_max(float v) {
#ifdef HAS_PL16
  v = fmaxf(v, pl16(v, 0x67452301u, 0xEFCDAB89u));  // xor 1
  v = fmaxf(v, pl16(v, 0x54761032u, 0xDCFE98BAu));  // xor 2
  v = fmaxf(v, pl16(v, 0x32107654u, 0xBA98FEDCu));  // xor 4
  v = fmaxf(v, pl16(v, 0xFEDCBA98u, 0x76543210u));  // xor 8
#else
#pragma unroll
  for (int m = 8; m >= 1; m >>= 1) v = fmaxf(v, __shfl_xor(v, m, 32));
#endif
  return v;
}
__device__ __forceinline__ float red16_sum(float v) {
#ifdef HAS_PL16
  v += pl16(v, 0x67452301u, 0xEFCDAB89u);
  v += pl16(v, 0x54761032u, 0xDCFE98BAu);
  v += pl16(v, 0x32107654u, 0xBA98FEDCu);
  v += pl16(v, 0xFEDCBA98u, 0x76543210u);
#else
#pragma unroll
  for (int m = 8; m >= 1; m >>= 1) v += __shfl_xor(v, m, 32);
#endif
  return v;
}

// ---- WMMA fragment loaders (layouts per ISA 7.12.2 / 7.12.4) --------------
__device__ __forceinline__ v16h frag_a_lds(const _Float16* base, int stride) {
  int lane = threadIdx.x & 31;
  int r = lane & 15, hh = lane >> 4;
  const _Float16* p = base + r * stride + hh * 8;
  union { v16h v; uint4 q[2]; } u;
  u.q[0] = *(const uint4*)(p);
  u.q[1] = *(const uint4*)(p + 16);
  return u.v;
}
__device__ __forceinline__ v16h frag_b_lds(const _Float16* base, int stride) {
  int lane = threadIdx.x & 31;
  int r = lane & 15, hh = lane >> 4;
  const _Float16* p = base + r * stride + hh * 8;
  union { v16h v; uint4 q[2]; } u;
  u.q[0] = *(const uint4*)(p);
  u.q[1] = *(const uint4*)(p + 16);
  return u.v;
}

// ---------------------------------------------------------------------------
// weight convert: f32 (rows x K) -> f16 (rows x Kp), zero-padded tail
// ---------------------------------------------------------------------------
__global__ __launch_bounds__(256) void convert_w(const float* __restrict__ src,
                                                 _Float16* __restrict__ dst,
                                                 int K, int Kp) {
  int row = blockIdx.x;
  for (int c = threadIdx.x; c < Kp; c += 256)
    dst[(size_t)row * Kp + c] = (c < K) ? (_Float16)src[(size_t)row * K + c]
                                        : (_Float16)0.f;
}

// ---------------------------------------------------------------------------
// Generic 32x256 GEMM block, double-buffered staging.
// ---------------------------------------------------------------------------
enum { MODE_BIAS = 0, MODE_PE_LN = 1, MODE_RESID_LN = 2, MODE_RELU = 3, MODE_RESID = 4 };

#define GA_STR 40
#define GB_STR 40

template <int MODE, typename InT>
__global__ __launch_bounds__(256) void gemm256(
    const InT* __restrict__ in, int K,            // K == A row stride
    const _Float16* __restrict__ W, int Kw,       // padded W stride (32x)
    const float* __restrict__ bias,
    const _Float16* __restrict__ aux,
    const float* __restrict__ gamma,
    const float* __restrict__ beta,
    _Float16* __restrict__ out)
{
  constexpr int ABYTES = 32 * GA_STR * 2;    // 2560
  constexpr int BBYTES = 256 * GB_STR * 2;   // 20480
  constexpr int LOOPB  = 2 * (ABYTES + BBYTES);           // 46080
  constexpr int SMEMB  = (LOOPB > 32 * 256 * 4) ? LOOPB : 32 * 256 * 4;
  __shared__ __align__(16) char smem[SMEMB];
  float* Clds = (float*)smem;                 // epilogue aliases loop buffers

  const int tid  = threadIdx.x;
  const int lane = tid & 31, w = tid >> 5;
  const int wr = w >> 2, wc = w & 3;
  const int mb = blockIdx.x * 32;
  const unsigned smb = lds_off_u32(smem);
  constexpr bool F16IN = (sizeof(InT) == 2);
  constexpr int GCNT = F16IN ? 5 : 4;         // async instrs per wave per tile

  const int arow  = tid >> 3;        // 0..31 (float-A path)
  const int acol0 = (tid & 7) * 4;
  const int ar8 = tid >> 3, as8 = tid & 7;    // f16-A path: one b64/lane

  union AF { uint2 u; _Float16 e[4]; };
  AF apre;

  auto issueA16 = [&](int k0, int buf) {
    async_ld_b64(smb + buf * ABYTES + ar8 * (GA_STR * 2) + as8 * 8,
                 (unsigned)(((mb + ar8) * K + k0 + as8 * 4) * 2), in);
  };
  auto loadAf = [&](int k0) {
    const InT* ap = in + (size_t)(mb + arow) * K + k0 + acol0;
#pragma unroll
    for (int i = 0; i < 4; ++i)
      apre.e[i] = (_Float16)((k0 + acol0 + i < K) ? (float)ap[i] : 0.f);
  };
  auto storeAf = [&](int buf) {
    *(uint2*)(smem + buf * ABYTES + (arow * GA_STR + acol0) * 2) = apre.u;
  };
  auto issueB = [&](int k0, int buf) {
    unsigned base = smb + 2 * ABYTES + buf * BBYTES;
#pragma unroll
    for (int i = 0; i < 4; ++i) {
      int e = tid + 256 * i;
      int n = e >> 2, s4 = e & 3;
      async_ld_b128(base + n * (GB_STR * 2) + s4 * 16,
                    (unsigned)((n * Kw + k0 + s4 * 8) * 2), W);
    }
  };

  v8f acc[4] = {};
  const int nK = (K + 31) >> 5;

  // prologue: stage tile 0 into buffer 0
  if (F16IN) issueA16(0, 0); else { loadAf(0); storeAf(0); }
  issueB(0, 0);

  for (int kt = 0; kt < nK; ++kt) {
    const int cur = kt & 1, nxt = cur ^ 1;
    const bool more = (kt + 1 < nK);
    if (more) {                      // overlap transfer t+1 with compute t
      const int k1 = (kt + 1) * 32;
      if (F16IN) issueA16(k1, nxt); else loadAf(k1);
      issueB(k1, nxt);
    }
    if (more) wait_async_le<GCNT>(); else wait_async_le<0>();
    __syncthreads();

    const _Float16* Ac = (const _Float16*)(smem + cur * ABYTES);
    const _Float16* Bc = (const _Float16*)(smem + 2 * ABYTES + cur * BBYTES);
    v16h af = frag_a_lds(Ac + wr * 16 * GA_STR, GA_STR);
#pragma unroll
    for (int t = 0; t < 4; ++t) {
      v16h bf = frag_b_lds(Bc + (wc * 64 + t * 16) * GB_STR, GB_STR);
      acc[t] = wmma_f16(af, bf, acc[t]);
    }
    if (more && !F16IN) storeAf(nxt);
    __syncthreads();
  }

  const int ci = lane & 15, hh = lane >> 4;

  if (MODE == MODE_BIAS || MODE == MODE_RELU || MODE == MODE_RESID) {
#pragma unroll
    for (int t = 0; t < 4; ++t) {
#pragma unroll
      for (int j = 0; j < 8; ++j) {
        int row = mb + wr * 16 + j + hh * 8;
        int col = wc * 64 + t * 16 + ci;
        float vv = acc[t][j] + bias[col];
        if (MODE == MODE_RELU)  vv = fmaxf(vv, 0.f);
        if (MODE == MODE_RESID) vv += (float)aux[(size_t)row * 256 + col];
        out[(size_t)row * 256 + col] = (_Float16)vv;
      }
    }
  } else {
#pragma unroll
    for (int t = 0; t < 4; ++t) {
#pragma unroll
      for (int j = 0; j < 8; ++j) {
        int rloc = wr * 16 + j + hh * 8;
        int row  = mb + rloc;
        int col  = wc * 64 + t * 16 + ci;
        float vv = acc[t][j] + bias[col];
        if (MODE == MODE_PE_LN) {
          int tpos = row & (SEQ - 1);
          float i2  = (float)(col & ~1);
          float div = __expf(-9.210340371976184f * i2 * (1.0f / 256.0f));
          float ang = (float)tpos * div;
          vv += (col & 1) ? __cosf(ang) : __sinf(ang);
        } else {
          vv += (float)aux[(size_t)row * 256 + col];
        }
        Clds[rloc * 256 + col] = vv;
      }
    }
    __syncthreads();
    for (int rr = w * 4; rr < w * 4 + 4; ++rr) {
      float s = 0.f, s2 = 0.f;
      float vals[8];
#pragma unroll
      for (int i = 0; i < 8; ++i) {
        float xv = Clds[rr * 256 + lane + 32 * i];
        vals[i] = xv; s += xv; s2 += xv * xv;
      }
#pragma unroll
      for (int m = 16; m >= 1; m >>= 1) {
        s  += __shfl_xor(s, m, 32);
        s2 += __shfl_xor(s2, m, 32);
      }
      float mean = s * (1.f / 256.f);
      float var  = s2 * (1.f / 256.f) - mean * mean;
      float rstd = rsqrtf(var + 1e-5f);
      size_t ro = (size_t)(mb + rr) * 256;
#pragma unroll
      for (int i = 0; i < 8; ++i) {
        int c = lane + 32 * i;
        out[ro + c] = (_Float16)((vals[i] - mean) * rstd * gamma[c] + beta[c]);
      }
    }
  }
}

// ---------------------------------------------------------------------------
// Flash-style cross attention, double-buffered K (async) / V (reg-prefetch).
// exp2-domain online softmax: q pre-scaled by hd^-0.5 * log2(e).
// ---------------------------------------------------------------------------
#define KSTR 72
#define PSTR 72
#define KBB  (64 * KSTR * 2)   // 9216 bytes per K/V buffer

__global__ __launch_bounds__(256) void attn_kernel(
    const _Float16* __restrict__ qg, const _Float16* __restrict__ kg,
    const _Float16* __restrict__ vg, float* __restrict__ ctx)
{
  __shared__ __align__(16) char asmem[4 * KBB + 8 * 16 * PSTR * 2];  // 55296 B
  const unsigned smb = lds_off_u32(asmem);

  const int tid = threadIdx.x, lane = tid & 31, w = tid >> 5;
  const int idx = blockIdx.x;
  const int b = idx >> 4, rem = idx & 15;
  const int head = rem >> 2, qt = rem & 3;
  const int qr0 = qt * 128 + w * 16;
  const int r = lane & 15, hh = lane >> 4;

  // q fragments pre-scaled into exp2 domain: 0.125 * log2(e)
  const float QS = 0.125f * 1.4426950408889634f;
  const _Float16* qrow = qg + ((size_t)(b * SEQ) + qr0 + r) * 256 + head * 64;
  v16h qf[2];
#pragma unroll
  for (int c = 0; c < 2; ++c) {
    union { v16h v; _Float16 e[16]; } u;
#pragma unroll
    for (int i = 0; i < 8; ++i) {
      int kk = c * 32 + hh * 8 + ((i < 4) ? 2 * i : 16 + 2 * (i - 4));
      u.e[2 * i]     = (_Float16)((float)qrow[kk]     * QS);
      u.e[2 * i + 1] = (_Float16)((float)qrow[kk + 1] * QS);
    }
    qf[c] = u.v;
  }

  union VU { uint4 u; _Float16 e16[8]; };
  VU vpre[2];

  auto issueK = [&](int kb, int buf) {
    unsigned base = smb + buf * KBB;
#pragma unroll
    for (int i = 0; i < 2; ++i) {
      int e = tid + 256 * i;
      int key = e >> 3, s8 = e & 7;
      async_ld_b128(base + key * (KSTR * 2) + s8 * 16,
                    (unsigned)((((b * SEQ) + kb + key) * 256 + head * 64 + s8 * 8) * 2),
                    kg);
    }
  };
  auto loadV = [&](int kb) {
#pragma unroll
    for (int i = 0; i < 2; ++i) {
      int e = tid + 256 * i;
      int hd = e >> 3, kg8 = e & 7;
#pragma unroll
      for (int j = 0; j < 8; ++j)
        vpre[i].e16[j] =
            vg[(size_t)((b * SEQ) + kb + kg8 * 8 + j) * 256 + head * 64 + hd];
    }
  };
  auto storeV = [&](int buf) {
    char* vb = asmem + 2 * KBB + buf * KBB;
#pragma unroll
    for (int i = 0; i < 2; ++i) {
      int e = tid + 256 * i;
      int hd = e >> 3, kg8 = e & 7;
      *(uint4*)(vb + (hd * KSTR + kg8 * 8) * 2) = vpre[i].u;
    }
  };

  v8f O[4] = {};
  float mrow[8], lrow[8];
#pragma unroll
  for (int j = 0; j < 8; ++j) { mrow[j] = -1e30f; lrow[j] = 0.f; }

  // prologue: stage key block 0
  issueK(0, 0); loadV(0); storeV(0);

  for (int t8 = 0; t8 < 8; ++t8) {
    const int cur = t8 & 1, nxt = cur ^ 1;
    const bool more = (t8 + 1 < 8);
    if (more) { issueK((t8 + 1) * 64, nxt); loadV((t8 + 1) * 64); }
    if (more) wait_async_le<2>(); else wait_async_le<0>();
    __syncthreads();

    const _Float16* Kc = (const _Float16*)(asmem + cur * KBB);
    const _Float16* Vc = (const _Float16*)(asmem + 2 * KBB + cur * KBB);
    _Float16* ps = (_Float16*)(asmem + 4 * KBB) + w * 16 * PSTR;

    // S = Q K^T for 64 keys (8 wmma)
    v8f s4[4];
#pragma unroll
    for (int nt = 0; nt < 4; ++nt) {
      v8f z = {};
      v16h b0 = frag_b_lds(Kc + nt * 16 * KSTR,      KSTR);
      v16h b1 = frag_b_lds(Kc + nt * 16 * KSTR + 32, KSTR);
      z = wmma_f16(qf[0], b0, z);
      z = wmma_f16(qf[1], b1, z);
      s4[nt] = z;
    }
    // online softmax (exp2 domain)
    float mnew[8], alpha[8];
#pragma unroll
    for (int j = 0; j < 8; ++j) {
      float t0 = fmaxf(fmaxf(s4[0][j], s4[1][j]), fmaxf(s4[2][j], s4[3][j]));
      t0 = red16_max(t0);
      mnew[j]  = fmaxf(mrow[j], t0);
      alpha[j] = fast_exp2(mrow[j] - mnew[j]);
    }
#pragma unroll
    for (int j = 0; j < 8; ++j) {
      float ts = 0.f;
#pragma unroll
      for (int nt = 0; nt < 4; ++nt) {
        float pp = fast_exp2(s4[nt][j] - mnew[j]);
        s4[nt][j] = pp;
        ts += pp;
      }
      ts = red16_sum(ts);
      lrow[j] = lrow[j] * alpha[j] + ts;
      mrow[j] = mnew[j];
    }
#pragma unroll
    for (int t = 0; t < 4; ++t)
#pragma unroll
      for (int j = 0; j < 8; ++j) O[t][j] *= alpha[j];

    // P (C-layout) -> A-layout via per-wave scratch; PV (8 wmma)
#pragma unroll
    for (int nt = 0; nt < 4; ++nt)
#pragma unroll
      for (int j = 0; j < 8; ++j)
        ps[(j + 8 * hh) * PSTR + nt * 16 + (lane & 15)] = (_Float16)s4[nt][j];
    asm volatile("s_wait_dscnt 0" ::: "memory");
#pragma unroll
    for (int kc = 0; kc < 64; kc += 32) {
      v16h pf = frag_a_lds(ps + kc, PSTR);
#pragma unroll
      for (int ht = 0; ht < 4; ++ht) {
        v16h bf = frag_b_lds(Vc + ht * 16 * KSTR + kc, KSTR);
        O[ht] = wmma_f16(pf, bf, O[ht]);
      }
    }
    if (more) storeV(nxt);
    __syncthreads();
  }

  float rl[8];
#pragma unroll
  for (int j = 0; j < 8; ++j) rl[j] = 1.f / lrow[j];
  size_t base = ((size_t)(b * SEQ) + qr0) * 256 + head * 64;
#pragma unroll
  for (int ht = 0; ht < 4; ++ht)
#pragma unroll
    for (int j = 0; j < 8; ++j)
      ctx[base + (size_t)(j + 8 * hh) * 256 + ht * 16 + (lane & 15)] = O[ht][j] * rl[j];
}

// ---------------------------------------------------------------------------
// mean over sequence + 2-way classifier
// ---------------------------------------------------------------------------
__global__ __launch_bounds__(256) void pool_cls_kernel(
    const _Float16* __restrict__ xm, const float* __restrict__ Wc,
    const float* __restrict__ bc, float* __restrict__ outp)
{
  __shared__ float pool[256];
  const int b = blockIdx.x, tid = threadIdx.x;
  float s = 0.f;
  for (int t = 0; t < SEQ; ++t) s += (float)xm[((size_t)(b * SEQ) + t) * 256 + tid];
  pool[tid] = s * (1.f / 512.f);
  __syncthreads();
  if (tid < 64) {
    int c = tid >> 5, lane = tid & 31;
    float acc = 0.f;
    for (int i = lane; i < 256; i += 32) acc += pool[i] * Wc[c * 256 + i];
#pragma unroll
    for (int m = 16; m >= 1; m >>= 1) acc += __shfl_xor(acc, m, 32);
    if (lane == 0) outp[b * 2 + c] = acc + bc[c];
  }
}

// ---------------------------------------------------------------------------
extern "C" void kernel_launch(void* const* d_in, const int* in_sizes, int n_in,
                              void* d_out, int out_size, void* d_ws, size_t ws_size,
                              hipStream_t stream)
{
  (void)in_sizes; (void)n_in; (void)out_size; (void)ws_size;
  const float* x   = (const float*)d_in[0];
  const float* y   = (const float*)d_in[1];
  /* d_in[2] = z, unused */
  const float* Wr  = (const float*)d_in[3];
  const float* br  = (const float*)d_in[4];
  const float* Wf  = (const float*)d_in[5];
  const float* bf  = (const float*)d_in[6];
  const float* g_x = (const float*)d_in[7];
  const float* b_x = (const float*)d_in[8];
  const float* g_y = (const float*)d_in[9];
  const float* b_y = (const float*)d_in[10];
  const float* Wq  = (const float*)d_in[11];
  const float* bq  = (const float*)d_in[12];
  const float* Wk  = (const float*)d_in[13];
  const float* bk  = (const float*)d_in[14];
  const float* Wv  = (const float*)d_in[15];
  const float* bv  = (const float*)d_in[16];
  const float* Wo  = (const float*)d_in[17];
  const float* bo  = (const float*)d_in[18];
  const float* g_m = (const float*)d_in[19];
  const float* b_m = (const float*)d_in[20];
  const float* W1  = (const float*)d_in[21];
  const float* b1  = (const float*)d_in[22];
  const float* W2  = (const float*)d_in[23];
  const float* b2  = (const float*)d_in[24];
  const float* Wc  = (const float*)d_in[25];
  const float* bc  = (const float*)d_in[26];

  char* p = (char*)d_ws;
  auto alloc = [&](size_t bytes) {
    char* r = p;
    p += (bytes + 255) & ~(size_t)255;
    return r;
  };
  _Float16* WrH = (_Float16*)alloc((size_t)256 * 800 * 2);
  _Float16* WfH = (_Float16*)alloc((size_t)256 * 704 * 2);
  _Float16* WqH = (_Float16*)alloc((size_t)256 * 256 * 2);
  _Float16* WkH = (_Float16*)alloc((size_t)256 * 256 * 2);
  _Float16* WvH = (_Float16*)alloc((size_t)256 * 256 * 2);
  _Float16* WoH = (_Float16*)alloc((size_t)256 * 256 * 2);
  _Float16* W1H = (_Float16*)alloc((size_t)256 * 256 * 2);
  _Float16* W2H = (_Float16*)alloc((size_t)256 * 256 * 2);
  const size_t HSLOT = (size_t)MTOT * 256 * 2;
  _Float16* xp = (_Float16*)alloc(HSLOT);
  _Float16* yp = (_Float16*)alloc(HSLOT);
  _Float16* qb = (_Float16*)alloc(HSLOT);
  _Float16* kb = (_Float16*)alloc(HSLOT);
  _Float16* vb = (_Float16*)alloc(HSLOT);
  float*    cx = (float*)alloc((size_t)MTOT * 256 * 4);
  _Float16* xr = kb;
  _Float16* h1 = qb;
  _Float16* xm = vb;

  dim3 blk(256);
  dim3 gM(MTOT / 32);
  dim3 gW(256);

  convert_w<<<gW, blk, 0, stream>>>(Wr, WrH, 769, 800);
  convert_w<<<gW, blk, 0, stream>>>(Wf, WfH, 674, 704);
  convert_w<<<gW, blk, 0, stream>>>(Wq, WqH, 256, 256);
  convert_w<<<gW, blk, 0, stream>>>(Wk, WkH, 256, 256);
  convert_w<<<gW, blk, 0, stream>>>(Wv, WvH, 256, 256);
  convert_w<<<gW, blk, 0, stream>>>(Wo, WoH, 256, 256);
  convert_w<<<gW, blk, 0, stream>>>(W1, W1H, 256, 256);
  convert_w<<<gW, blk, 0, stream>>>(W2, W2H, 256, 256);

  gemm256<MODE_PE_LN, float>    <<<gM, blk, 0, stream>>>(x,  769, WrH, 800, br, nullptr, g_x, b_x, xp);
  gemm256<MODE_PE_LN, float>    <<<gM, blk, 0, stream>>>(y,  674, WfH, 704, bf, nullptr, g_y, b_y, yp);
  gemm256<MODE_BIAS, _Float16>  <<<gM, blk, 0, stream>>>(xp, 256, WqH, 256, bq, nullptr, nullptr, nullptr, qb);
  gemm256<MODE_BIAS, _Float16>  <<<gM, blk, 0, stream>>>(yp, 256, WkH, 256, bk, nullptr, nullptr, nullptr, kb);
  gemm256<MODE_BIAS, _Float16>  <<<gM, blk, 0, stream>>>(yp, 256, WvH, 256, bv, nullptr, nullptr, nullptr, vb);
  attn_kernel                   <<<dim3(BATCH * NHEAD * 4), blk, 0, stream>>>(qb, kb, vb, cx);
  gemm256<MODE_RESID_LN, float> <<<gM, blk, 0, stream>>>(cx, 256, WoH, 256, bo, xp, g_m, b_m, xr);
  gemm256<MODE_RELU, _Float16>  <<<gM, blk, 0, stream>>>(xr, 256, W1H, 256, b1, nullptr, nullptr, nullptr, h1);
  gemm256<MODE_RESID, _Float16> <<<gM, blk, 0, stream>>>(h1, 256, W2H, 256, b2, xr, nullptr, nullptr, xm);
  pool_cls_kernel               <<<dim3(BATCH), blk, 0, stream>>>(xm, Wc, bc, (float*)d_out);
}
